// HSICLoss_81965155877487
// MI455X (gfx1250) — compile-verified, matrix operand order checked by reading, not run on hardware
//
#include <hip/hip_runtime.h>

typedef unsigned short u16;
typedef unsigned int   u32;

typedef __attribute__((ext_vector_type(16))) __bf16 v16bf;
typedef __attribute__((ext_vector_type(8)))  float  v8f;

#define NROWS 8192
#define DX    128
#define DY    64
#define NBLK  64          // 128-row blocks
#define NTILES 2080       // NBLK*(NBLK+1)/2 upper-triangle tile pairs
#define NWG    512

// ---------------- workspace layout (bytes) ----------------
#define WS_XHI   ((size_t)0)
#define WS_XLO   (WS_XHI + (size_t)NROWS*DX*2)
#define WS_YHI   (WS_XLO + (size_t)NROWS*DX*2)
#define WS_YLO   (WS_YHI + (size_t)NROWS*DY*2)
#define WS_SQX   (WS_YLO + (size_t)NROWS*DY*2)
#define WS_SQY   (WS_SQX + (size_t)NROWS*4)
#define WS_RSX   (WS_SQY + (size_t)NROWS*4)          // [64][8192] f32 = 2 MiB
#define WS_RSY   (WS_RSX + (size_t)NBLK*NROWS*4)
#define WS_PP    (WS_RSY + (size_t)NBLK*NROWS*4)     // [512] f32
#define WS_END   (WS_PP  + (size_t)NWG*4)

// ---------------- dynamic LDS layout for tile kernel ----------------
#define L_XHI_I  0
#define L_XLO_I  32768
#define L_YHI_I  65536
#define L_YLO_I  81920
#define L_XHI_J0 98304
#define L_XLO_J0 131072
#define L_YHI_J0 163840
#define L_YLO_J0 180224
#define L_XHI_J1 196608
#define L_XLO_J1 229376
#define L_YHI_J1 262144
#define L_YLO_J1 278528
#define L_SQX_I  294912
#define L_SQY_I  295424
#define L_SQX_J  295936   // [2][128] f32
#define L_SQY_J  296960   // [2][128] f32
#define L_RSX    297984   // [2][128] f32
#define L_RSY    299008
#define L_CSX    300032   // [4][128] f32
#define L_CSY    302080
#define L_PRED   304128   // 8 f32
#define L_TOTAL  304256

__device__ __forceinline__ u16 f2bf(float f) {
    u32 u = __float_as_uint(f);
    u32 r = u + 0x7FFFu + ((u >> 16) & 1u);   // round-to-nearest-even
    return (u16)(r >> 16);
}
__device__ __forceinline__ float bf2f(u16 h) {
    return __uint_as_float(((u32)h) << 16);
}

// ============ kernel 1: split fp32 -> bf16 hi/lo, exact fp32 norms ============
__global__ __launch_bounds__(128)
void hsic_prep(const float* __restrict__ x, const float* __restrict__ y,
               u16* __restrict__ xhi, u16* __restrict__ xlo,
               u16* __restrict__ yhi, u16* __restrict__ ylo,
               float* __restrict__ sqx, float* __restrict__ sqy) {
    __shared__ float sred[128];
    const int r = blockIdx.x, t = threadIdx.x;

    float f = x[(size_t)r*DX + t];
    u16 h = f2bf(f);
    xhi[(size_t)r*DX + t] = h;
    xlo[(size_t)r*DX + t] = f2bf(f - bf2f(h));
    sred[t] = f * f;
    __syncthreads();
    for (int s = 64; s > 0; s >>= 1) { if (t < s) sred[t] += sred[t + s]; __syncthreads(); }
    if (t == 0) sqx[r] = sred[0];
    __syncthreads();

    float g = 0.0f;
    if (t < DY) {
        g = y[(size_t)r*DY + t];
        u16 hy = f2bf(g);
        yhi[(size_t)r*DY + t] = hy;
        ylo[(size_t)r*DY + t] = f2bf(g - bf2f(hy));
    }
    sred[t] = (t < DY) ? g * g : 0.0f;
    __syncthreads();
    for (int s = 64; s > 0; s >>= 1) { if (t < s) sred[t] += sred[t + s]; __syncthreads(); }
    if (t == 0) sqy[r] = sred[0];
}

// ---------------- helpers for kernel 2 ----------------
__device__ __forceinline__ v8f wmma_bf16(v16bf a, v16bf b, v8f c) {
    return __builtin_amdgcn_wmma_f32_16x16x32_bf16(false, a, false, b, (short)0, c,
                                                   false, false);
}

template <int ROWB, int CM>
__device__ __forceinline__ v16bf ld_frag(const char* base, int row, int ks, int g) {
    const char* rp = base + row * ROWB;
    int sw = row & CM;
    int c0 = ks * 4 + g;       // K = ks*32 + 8*g
    int c1 = c0 + 2;           // K = ks*32 + 16 + 8*g
    union { uint4 u[2]; v16bf v; } u;
    u.u[0] = *(const uint4*)(rp + ((c0 ^ sw) << 4));
    u.u[1] = *(const uint4*)(rp + ((c1 ^ sw) << 4));
    return u.v;
}

// synchronous staging (I block): global -> VGPR -> swizzled LDS
__device__ __forceinline__ void stage_x(char* dst, const u16* src, int row0, int tid) {
#pragma unroll
    for (int it = 0; it < 8; ++it) {
        int id = tid + it * 256;
        int r = id >> 4, c = id & 15;
        uint4 v = *(const uint4*)((const char*)src + (size_t)(row0 + r) * 256 + (c << 4));
        *(uint4*)(dst + r * 256 + (((c ^ (r & 15)) << 4))) = v;
    }
}
__device__ __forceinline__ void stage_y(char* dst, const u16* src, int row0, int tid) {
#pragma unroll
    for (int it = 0; it < 4; ++it) {
        int id = tid + it * 256;
        int r = id >> 3, c = id & 7;
        uint4 v = *(const uint4*)((const char*)src + (size_t)(row0 + r) * 128 + (c << 4));
        *(uint4*)(dst + r * 128 + (((c ^ (r & 7)) << 4))) = v;
    }
}

// asynchronous staging (J block): GLOBAL_LOAD_ASYNC_TO_LDS_B128, ASYNCcnt-tracked
__device__ __forceinline__ void a_stage_x(u32 ldsbase, const u16* src, int row0, int tid) {
#pragma unroll
    for (int it = 0; it < 8; ++it) {
        int id = tid + it * 256;
        int r = id >> 4, c = id & 15;
        u32 goff = (u32)(((row0 + r) << 8) + (c << 4));
        u32 loff = ldsbase + (u32)((r << 8) + (((c ^ (r & 15)) << 4)));
        asm volatile("global_load_async_to_lds_b128 %0, %1, %2"
                     :: "v"(loff), "v"(goff), "s"(src) : "memory");
    }
}
__device__ __forceinline__ void a_stage_y(u32 ldsbase, const u16* src, int row0, int tid) {
#pragma unroll
    for (int it = 0; it < 4; ++it) {
        int id = tid + it * 256;
        int r = id >> 3, c = id & 7;
        u32 goff = (u32)(((row0 + r) << 7) + (c << 4));
        u32 loff = ldsbase + (u32)((r << 7) + (((c ^ (r & 7)) << 4)));
        asm volatile("global_load_async_to_lds_b128 %0, %1, %2"
                     :: "v"(loff), "v"(goff), "s"(src) : "memory");
    }
}
__device__ __forceinline__ void wait_async0() {
    asm volatile("s_wait_asynccnt 0x0" ::: "memory");
}

// decode linear upper-triangle tile index -> (bi, bj), bj >= bi
__device__ __forceinline__ void tri_decode(int t, int& bi, int& bj) {
    int r = (int)(64.5f - sqrtf(64.5f * 64.5f - 2.0f * (float)t));
    if (r < 0) r = 0;
    while (r * (129 - r) / 2 > t) --r;
    while ((r + 1) * (129 - (r + 1)) / 2 <= t) ++r;
    bi = r;
    bj = r + (t - r * (129 - r) / 2);
}

// ============ kernel 2: symmetric tiled WMMA Gram + streaming reductions ============
__global__ __launch_bounds__(256)
void hsic_tile(const u16* __restrict__ xhi, const u16* __restrict__ xlo,
               const u16* __restrict__ yhi, const u16* __restrict__ ylo,
               const float* __restrict__ sqx, const float* __restrict__ sqy,
               float* __restrict__ rsxPart, float* __restrict__ rsyPart,
               float* __restrict__ pPart) {
    extern __shared__ char smem[];
    float* ssqxI = (float*)(smem + L_SQX_I);
    float* ssqyI = (float*)(smem + L_SQY_I);
    float* ssqxJb = (float*)(smem + L_SQX_J);   // [2][128]
    float* ssqyJb = (float*)(smem + L_SQY_J);
    float* rsx_s = (float*)(smem + L_RSX);      // [2][128]
    float* rsy_s = (float*)(smem + L_RSY);
    float* csx_s = (float*)(smem + L_CSX);      // [4][128]
    float* csy_s = (float*)(smem + L_CSY);
    float* pred  = (float*)(smem + L_PRED);
    const u32 smb = (u32)(size_t)smem;          // low 32 bits = LDS byte offset

    const int tid  = threadIdx.x;
    const int lane = tid & 31;
    const int wave = tid >> 5;
    const int g    = lane >> 4;
    const int ln   = lane & 15;
    const int wm   = wave & 3;       // rows wm*32..+32
    const int wn   = wave >> 2;      // cols wn*64..+64
    const float KF = -0.7213475204444817f;   // -log2(e)/2 (sigma = 1)

    // contiguous segment of the flat upper-triangle tile list
    const int t0 = (blockIdx.x * NTILES) / NWG;
    const int t1 = ((blockIdx.x + 1) * NTILES) / NWG;

    int cbi, cbj;
    tri_decode(t0, cbi, cbj);

    // stage I block (sync) and first J tile (async, buffer 0)
    stage_x(smem + L_XHI_I, xhi, cbi * 128, tid);
    stage_x(smem + L_XLO_I, xlo, cbi * 128, tid);
    stage_y(smem + L_YHI_I, yhi, cbi * 128, tid);
    stage_y(smem + L_YLO_I, ylo, cbi * 128, tid);
    if (tid < 128) { ssqxI[tid] = sqx[cbi * 128 + tid]; ssqyI[tid] = sqy[cbi * 128 + tid]; }
    a_stage_x(smb + L_XHI_J0, xhi, cbj * 128, tid);
    a_stage_x(smb + L_XLO_J0, xlo, cbj * 128, tid);
    a_stage_y(smb + L_YHI_J0, yhi, cbj * 128, tid);
    a_stage_y(smb + L_YLO_J0, ylo, cbj * 128, tid);
    if (tid < 128) { ssqxJb[tid] = sqx[cbj * 128 + tid]; ssqyJb[tid] = sqy[cbj * 128 + tid]; }

    float p = 0.0f;
    int cur = 0;
    int nbi = cbi, nbj = cbj;

    for (int t = t0; t < t1; ++t) {
        const bool have_next = (t + 1 < t1);
        if (have_next) {                       // next tile coordinates
            nbj = cbj + 1; nbi = cbi;
            if (nbj >= NBLK) { nbi = cbi + 1; nbj = nbi; }
        }

        wait_async0();                         // my async copies into buf `cur` done
        __syncthreads();                       // everyone's copies done; prior reads done

        // prefetch next J block into the other buffer (overlaps with compute below)
        if (have_next) {
            const u32 jb = cur ? (u32)L_XHI_J0 : (u32)L_XHI_J1;  // other buffer base
            a_stage_x(smb + jb,                        xhi, nbj * 128, tid);
            a_stage_x(smb + jb + (L_XLO_J0 - L_XHI_J0), xlo, nbj * 128, tid);
            a_stage_y(smb + jb + (L_YHI_J0 - L_XHI_J0), yhi, nbj * 128, tid);
            a_stage_y(smb + jb + (L_YLO_J0 - L_XHI_J0), ylo, nbj * 128, tid);
            if (tid < 128) {
                ssqxJb[(1 - cur) * 128 + tid] = sqx[nbj * 128 + tid];
                ssqyJb[(1 - cur) * 128 + tid] = sqy[nbj * 128 + tid];
            }
        }

        const char* xhiJ = smem + (cur ? L_XHI_J1 : L_XHI_J0);
        const char* xloJ = smem + (cur ? L_XLO_J1 : L_XLO_J0);
        const char* yhiJ = smem + (cur ? L_YHI_J1 : L_YHI_J0);
        const char* yloJ = smem + (cur ? L_YLO_J1 : L_YLO_J0);
        const float* ssqxJ = ssqxJb + cur * 128;
        const float* ssqyJ = ssqyJb + cur * 128;
        const bool diag = (cbi == cbj);
        const float wt = diag ? 1.0f : 2.0f;

        // -------- X Gram block (split bf16, K=128) --------
        v8f acc[2][4];
#pragma unroll
        for (int mt = 0; mt < 2; ++mt)
#pragma unroll
            for (int nt = 0; nt < 4; ++nt) acc[mt][nt] = (v8f)(0.0f);
#pragma unroll
        for (int ks = 0; ks < 4; ++ks) {
            v16bf ah[2], al[2], bh[4], bl[4];
#pragma unroll
            for (int mt = 0; mt < 2; ++mt) {
                int r = wm * 32 + mt * 16 + ln;
                ah[mt] = ld_frag<256, 15>(smem + L_XHI_I, r, ks, g);
                al[mt] = ld_frag<256, 15>(smem + L_XLO_I, r, ks, g);
            }
#pragma unroll
            for (int nt = 0; nt < 4; ++nt) {
                int r = wn * 64 + nt * 16 + ln;
                bh[nt] = ld_frag<256, 15>(xhiJ, r, ks, g);
                bl[nt] = ld_frag<256, 15>(xloJ, r, ks, g);
            }
#pragma unroll
            for (int mt = 0; mt < 2; ++mt)
#pragma unroll
                for (int nt = 0; nt < 4; ++nt) {
                    acc[mt][nt] = wmma_bf16(ah[mt], bh[nt], acc[mt][nt]);
                    acc[mt][nt] = wmma_bf16(ah[mt], bl[nt], acc[mt][nt]);
                    acc[mt][nt] = wmma_bf16(al[mt], bh[nt], acc[mt][nt]);
                }
        }
#pragma unroll
        for (int mt = 0; mt < 2; ++mt)
#pragma unroll
            for (int nt = 0; nt < 4; ++nt)
#pragma unroll
                for (int e = 0; e < 8; ++e) {
                    int m = wm * 32 + mt * 16 + g * 8 + e;
                    int n = wn * 64 + nt * 16 + ln;
                    float d2 = ssqxI[m] + ssqxJ[n] - 2.0f * acc[mt][nt][e];
                    acc[mt][nt][e] = __builtin_amdgcn_exp2f(fmaxf(d2, 0.0f) * KF);
                }

        // -------- Y Gram block (K=64) --------
        v8f accy[2][4];
#pragma unroll
        for (int mt = 0; mt < 2; ++mt)
#pragma unroll
            for (int nt = 0; nt < 4; ++nt) accy[mt][nt] = (v8f)(0.0f);
#pragma unroll
        for (int ks = 0; ks < 2; ++ks) {
            v16bf ah[2], al[2], bh[4], bl[4];
#pragma unroll
            for (int mt = 0; mt < 2; ++mt) {
                int r = wm * 32 + mt * 16 + ln;
                ah[mt] = ld_frag<128, 7>(smem + L_YHI_I, r, ks, g);
                al[mt] = ld_frag<128, 7>(smem + L_YLO_I, r, ks, g);
            }
#pragma unroll
            for (int nt = 0; nt < 4; ++nt) {
                int r = wn * 64 + nt * 16 + ln;
                bh[nt] = ld_frag<128, 7>(yhiJ, r, ks, g);
                bl[nt] = ld_frag<128, 7>(yloJ, r, ks, g);
            }
#pragma unroll
            for (int mt = 0; mt < 2; ++mt)
#pragma unroll
                for (int nt = 0; nt < 4; ++nt) {
                    accy[mt][nt] = wmma_bf16(ah[mt], bh[nt], accy[mt][nt]);
                    accy[mt][nt] = wmma_bf16(ah[mt], bl[nt], accy[mt][nt]);
                    accy[mt][nt] = wmma_bf16(al[mt], bh[nt], accy[mt][nt]);
                }
        }

        // -------- combine --------
        float ptile = 0.0f;
        float rsxp[16], rsyp[16], cxp[4], cyp[4];
#pragma unroll
        for (int i = 0; i < 16; ++i) { rsxp[i] = 0.0f; rsyp[i] = 0.0f; }
#pragma unroll
        for (int i = 0; i < 4; ++i)  { cxp[i] = 0.0f;  cyp[i] = 0.0f; }

#pragma unroll
        for (int mt = 0; mt < 2; ++mt)
#pragma unroll
            for (int nt = 0; nt < 4; ++nt)
#pragma unroll
                for (int e = 0; e < 8; ++e) {
                    int m = wm * 32 + mt * 16 + g * 8 + e;
                    int n = wn * 64 + nt * 16 + ln;
                    float d2 = ssqyI[m] + ssqyJ[n] - 2.0f * accy[mt][nt][e];
                    float ky = __builtin_amdgcn_exp2f(fmaxf(d2, 0.0f) * KF);
                    float kx = acc[mt][nt][e];
                    ptile += kx * ky;
                    rsxp[mt * 8 + e] += kx;  rsyp[mt * 8 + e] += ky;
                    cxp[nt] += kx;           cyp[nt] += ky;
                }
        p += wt * ptile;

        // row-sum reduce (over n within tile): lanes of same g-group share rows
#pragma unroll
        for (int i = 0; i < 16; ++i) {
            float vx = rsxp[i], vy = rsyp[i];
            vx += __shfl_xor(vx, 1, 32); vx += __shfl_xor(vx, 2, 32);
            vx += __shfl_xor(vx, 4, 32); vx += __shfl_xor(vx, 8, 32);
            vy += __shfl_xor(vy, 1, 32); vy += __shfl_xor(vy, 2, 32);
            vy += __shfl_xor(vy, 4, 32); vy += __shfl_xor(vy, 8, 32);
            if (ln == 0) {
                int m = wm * 32 + (i >> 3) * 16 + (i & 7) + 8 * g;
                rsx_s[wn * 128 + m] = vx;
                rsy_s[wn * 128 + m] = vy;
            }
        }
        // col-sum reduce (over m within tile): lanes L, L^16 share columns
#pragma unroll
        for (int i = 0; i < 4; ++i) {
            float vx = cxp[i], vy = cyp[i];
            vx += __shfl_xor(vx, 16, 32);
            vy += __shfl_xor(vy, 16, 32);
            if (g == 0) {
                int n = wn * 64 + i * 16 + ln;
                csx_s[wm * 128 + n] = vx;
                csy_s[wm * 128 + n] = vy;
            }
        }
        __syncthreads();
        if (tid < 128) {
            // row sums of tile (bi,bj) -> rows of block bi, partial slot bj
            rsxPart[(size_t)cbj * NROWS + cbi * 128 + tid] = rsx_s[tid] + rsx_s[128 + tid];
            rsyPart[(size_t)cbj * NROWS + cbi * 128 + tid] = rsy_s[tid] + rsy_s[128 + tid];
            if (!diag) {  // col sums -> rows of block bj (symmetry), partial slot bi
                float cx = csx_s[tid] + csx_s[128 + tid] + csx_s[256 + tid] + csx_s[384 + tid];
                float cy = csy_s[tid] + csy_s[128 + tid] + csy_s[256 + tid] + csy_s[384 + tid];
                rsxPart[(size_t)cbi * NROWS + cbj * 128 + tid] = cx;
                rsyPart[(size_t)cbi * NROWS + cbj * 128 + tid] = cy;
            }
        }

        // restage I block if the next tile starts a new row of blocks
        if (have_next && nbi != cbi) {
            __syncthreads();   // all waves done reading I buffers
            stage_x(smem + L_XHI_I, xhi, nbi * 128, tid);
            stage_x(smem + L_XLO_I, xlo, nbi * 128, tid);
            stage_y(smem + L_YHI_I, yhi, nbi * 128, tid);
            stage_y(smem + L_YLO_I, ylo, nbi * 128, tid);
            if (tid < 128) {
                ssqxI[tid] = sqx[nbi * 128 + tid];
                ssqyI[tid] = sqy[nbi * 128 + tid];
            }
        }
        cbi = nbi; cbj = nbj;
        cur ^= 1;
    }

    // weighted elementwise-product partial for this WG
    p += __shfl_xor(p, 1, 32);  p += __shfl_xor(p, 2, 32);
    p += __shfl_xor(p, 4, 32);  p += __shfl_xor(p, 8, 32);
    p += __shfl_xor(p, 16, 32);
    if (lane == 0) pred[wave] = p;
    __syncthreads();
    if (tid == 0) {
        float pp = 0.0f;
#pragma unroll
        for (int w = 0; w < 8; ++w) pp += pred[w];
        pPart[blockIdx.x] = pp;
    }
}

// ============ kernel 3: deterministic final combine ============
__global__ __launch_bounds__(256)
void hsic_final(const float* __restrict__ rsxPart, const float* __restrict__ rsyPart,
                const float* __restrict__ pPart, float* __restrict__ out) {
    __shared__ float s0[256], s1[256], s2[256], s3[256];
    const int t = threadIdx.x;
    float t1 = 0.0f, t2 = 0.0f, sx = 0.0f, sy = 0.0f;
    for (int i = t; i < NROWS; i += 256) {
        float a = 0.0f, b = 0.0f;
        for (int k = 0; k < NBLK; ++k) {
            a += rsxPart[(size_t)k * NROWS + i];
            b += rsyPart[(size_t)k * NROWS + i];
        }
        sx += a; sy += b; t2 += a * b;
    }
    for (int i = t; i < NWG; i += 256) t1 += pPart[i];
    s0[t] = t1; s1[t] = t2; s2[t] = sx; s3[t] = sy;
    __syncthreads();
    for (int s = 128; s > 0; s >>= 1) {
        if (t < s) { s0[t] += s0[t+s]; s1[t] += s1[t+s]; s2[t] += s2[t+s]; s3[t] += s3[t+s]; }
        __syncthreads();
    }
    if (t == 0) {
        const float invN = 1.0f / (float)NROWS;
        out[0] = s0[0] - 2.0f * invN * s1[0] + (s2[0] * invN) * (s3[0] * invN);
    }
}

// ============ host-side launcher ============
extern "C" void kernel_launch(void* const* d_in, const int* in_sizes, int n_in,
                              void* d_out, int out_size, void* d_ws, size_t ws_size,
                              hipStream_t stream) {
    const float* x = (const float*)d_in[0];   // [8192,128] f32
    const float* y = (const float*)d_in[1];   // [8192, 64] f32
    float* out = (float*)d_out;
    char*  ws  = (char*)d_ws;

    u16*   xhi = (u16*)(ws + WS_XHI);
    u16*   xlo = (u16*)(ws + WS_XLO);
    u16*   yhi = (u16*)(ws + WS_YHI);
    u16*   ylo = (u16*)(ws + WS_YLO);
    float* sqx = (float*)(ws + WS_SQX);
    float* sqy = (float*)(ws + WS_SQY);
    float* rsx = (float*)(ws + WS_RSX);
    float* rsy = (float*)(ws + WS_RSY);
    float* pp  = (float*)(ws + WS_PP);

    static bool attr_set = false;
    if (!attr_set) {
        hipFuncSetAttribute((const void*)hsic_tile,
                            hipFuncAttributeMaxDynamicSharedMemorySize, L_TOTAL);
        attr_set = true;
    }

    hipLaunchKernelGGL(hsic_prep, dim3(NROWS), dim3(128), 0, stream,
                       x, y, xhi, xlo, yhi, ylo, sqx, sqy);

    hipLaunchKernelGGL(hsic_tile, dim3(NWG), dim3(256), L_TOTAL, stream,
                       xhi, xlo, yhi, ylo, sqx, sqy, rsx, rsy, pp);

    hipLaunchKernelGGL(hsic_final, dim3(1), dim3(256), 0, stream,
                       rsx, rsy, pp, out);
}
